// MDTAWithStrategy_27728308863243
// MI455X (gfx1250) — matrix operand order repaired and checked by reading, not compiled, not run on hardware
//
#include <hip/hip_runtime.h>
#include <hip/hip_bf16.h>
#include <stdint.h>

// ---------------- problem constants ----------------
constexpr int B_    = 2;
constexpr int N_X   = 4096;
constexpr int N_TOT = 4104;   // 4096 + 8 strategy tokens
constexpr int N_PAD = 4128;   // padded to multiple of 32 for WMMA K loops
constexpr int C_    = 192;
constexpr int C3    = 576;
constexpr int H_    = 8;
constexpr int D_    = 24;
constexpr int DP    = 32;     // padded head dim (zero-filled 24..31)

constexpr size_t XIN_ELEMS  = (size_t)B_ * N_PAD * C_;      // bf16
constexpr size_t WQKV_ELEMS = (size_t)C3 * C_;              // bf16
constexpr size_t WOUT_ELEMS = (size_t)C_ * C_;              // bf16
constexpr size_t QKV_ELEMS  = (size_t)B_ * H_ * N_PAD * DP; // bf16 (each of Q,K,Vt)

typedef __attribute__((ext_vector_type(16))) __bf16 v16bf;
typedef __attribute__((ext_vector_type(8)))  __bf16 v8bf;
typedef __attribute__((ext_vector_type(8)))  float  v8f;

// D = A*B + C, bf16 inputs, f32 accumulate (v_wmma_f32_16x16x32_bf16)
static __device__ __forceinline__ v8f wmma_bf16(v16bf a, v16bf b, v8f c) {
  return __builtin_amdgcn_wmma_f32_16x16x32_bf16(
      /*neg_a=*/false, a, /*neg_b=*/false, b,
      /*c_mod=*/(short)0, c, /*reuse_a=*/false, /*reuse_b=*/false);
}

// A fragment (16x32, M x K), source row-major with leading dim ld (elements).
// ISA layout: lane L: M = L&15, half = L>>4; elem e: K = half*8 + e + (e>=8 ? 8 : 0)
static __device__ __forceinline__ v16bf load_A(const __bf16* p, int ld, int lane) {
  int m = lane & 15, qh = lane >> 4;
  const __bf16* r = p + (size_t)m * ld + qh * 8;
  v8bf lo = *(const v8bf*)(r);        // K = qh*8 + 0..7
  v8bf hi = *(const v8bf*)(r + 16);   // K = 16 + qh*8 + 0..7
  v16bf out;
#pragma unroll
  for (int i = 0; i < 8; i++) { out[i] = lo[i]; out[i + 8] = hi[i]; }
  return out;
}

// B fragment (32x16, K x N), source stored as 16 rows (one per N) of K-contiguous
// bf16 with leading dim ld. ISA layout: lane L: N = L&15, half = L>>4; K = half*8 + e + (e>=8 ? 8 : 0)
static __device__ __forceinline__ v16bf load_B(const __bf16* p, int ld, int lane) {
  int n = lane & 15, qh = lane >> 4;
  const __bf16* r = p + (size_t)n * ld + qh * 8;
  v8bf lo = *(const v8bf*)(r);
  v8bf hi = *(const v8bf*)(r + 16);
  v16bf out;
#pragma unroll
  for (int i = 0; i < 8; i++) { out[i] = lo[i]; out[i + 8] = hi[i]; }
  return out;
}

// ---------------- pack / init kernels ----------------
__global__ void k_pack_x(const float* __restrict__ X, const float* __restrict__ S,
                         __bf16* __restrict__ xin) {
  size_t i = (size_t)blockIdx.x * 256 + threadIdx.x;
  if (i >= XIN_ELEMS) return;
  int c = (int)(i % C_);
  size_t t = i / C_;
  int n = (int)(t % N_PAD);
  int b = (int)(t / N_PAD);
  float v = 0.0f;
  if (n < N_X)        v = X[((size_t)b * N_X + n) * C_ + c];
  else if (n < N_TOT) v = S[((size_t)b * 8 + (n - N_X)) * C_ + c];
  xin[i] = (__bf16)v;
}

__global__ void k_pack_w(const float* __restrict__ wqkv, const float* __restrict__ wout,
                         __bf16* __restrict__ dq, __bf16* __restrict__ dw) {
  int i = blockIdx.x * 256 + threadIdx.x;
  if (i < (int)WQKV_ELEMS) dq[i] = (__bf16)wqkv[i];
  else {
    int j = i - (int)WQKV_ELEMS;
    if (j < (int)WOUT_ELEMS) dw[j] = (__bf16)wout[j];
  }
}

__global__ void k_zero(uint32_t* __restrict__ p, size_t n32) {
  size_t i = (size_t)blockIdx.x * 256 + threadIdx.x;
  if (i < n32) p[i] = 0u;
}

// ---------------- QKV projection GEMM ----------------
// qkv[row, o] = sum_c Xin[row, c] * Wqkv[o, c]; one wave per 16x16 tile.
// grid = (C3/16, B*N_PAD/16). Scatter into Q[b][h][n][dp], K[b][h][n][dp],
// Vt[b][h][dp][n] (bf16, dp 24..31 stays 0 from k_zero)
__global__ __launch_bounds__(32) void k_qkv(const __bf16* __restrict__ xin,
                                            const __bf16* __restrict__ wqkv,
                                            __bf16* __restrict__ Q,
                                            __bf16* __restrict__ K,
                                            __bf16* __restrict__ Vt) {
  int otile = blockIdx.x;
  int ntile = blockIdx.y;
  int lane  = threadIdx.x;
  int row0  = ntile * 16;   // row over B*N_PAD; a 16-row tile never crosses a batch
  int o0    = otile * 16;

  const __bf16* a0 = xin  + (size_t)row0 * C_;
  const __bf16* b0 = wqkv + (size_t)o0 * C_;
  v8f acc = {};
#pragma unroll
  for (int k = 0; k < C_; k += 32) {
    v16bf a = load_A(a0 + k, C_, lane);
    v16bf b = load_B(b0 + k, C_, lane);
    acc = wmma_bf16(a, b, acc);
  }

  int nl = lane & 15, qh = lane >> 4;
  int o   = o0 + nl;
  int sec = o / C_;        // 0=q 1=k 2=v
  int oc  = o % C_;
  int h   = oc / D_;
  int d   = oc % D_;
  int b_i   = row0 / N_PAD;          // scalar (uniform over wave)
  int nbase = row0 - b_i * N_PAD;
  const size_t bh = (size_t)b_i * H_ + h;
#pragma unroll
  for (int r = 0; r < 8; r++) {
    int n = nbase + r + 8 * qh;
    __bf16 val = (__bf16)acc[r];
    if (sec == 2) {
      Vt[(bh * DP + d) * N_PAD + n] = val;
    } else {
      size_t idx = (bh * N_PAD + n) * DP + d;
      if (sec == 0) Q[idx] = val; else K[idx] = val;
    }
  }
}

// ---------------- flash attention ----------------
// One wave per (b, h, 32-row query tile pair): two 16-row M-tiles share every
// K/V fragment, so 8 WMMAs are issued per 4 fragment loads.
// grid = (N_PAD/32, B*H)
__global__ __launch_bounds__(32) void k_attn(const __bf16* __restrict__ Q,
                                             const __bf16* __restrict__ K,
                                             const __bf16* __restrict__ Vt,
                                             const float* __restrict__ temperature,
                                             __bf16* __restrict__ AO) {
  __shared__ __attribute__((aligned(16))) __bf16 Plds[32 * 32];

  int it = blockIdx.x;           // 0..128 (32-row tile)
  int bh = blockIdx.y;           // 0..15
  int b = bh >> 3, h = bh & 7;
  int lane = threadIdx.x;
  int nl = lane & 15, qh = lane >> 4;

  const __bf16* Qb = Q  + ((size_t)b * H_ + h) * (size_t)N_PAD * DP;
  const __bf16* Kb = K  + ((size_t)b * H_ + h) * (size_t)N_PAD * DP;
  const __bf16* Vb = Vt + ((size_t)b * H_ + h) * (size_t)DP * N_PAD;
  float tscale = temperature[h];

  // Q fragments for the two 16-row tiles (K-dim = DP = 32 -> one fragment each)
  v16bf qa0 = load_A(Qb + (size_t)(it * 32) * DP, DP, lane);
  v16bf qa1 = load_A(Qb + (size_t)(it * 32 + 16) * DP, DP, lane);

  v8f o00 = {}, o01 = {};   // tile0: d 0..15, d 16..31
  v8f o10 = {}, o11 = {};   // tile1
  float mrun[2][8], lrun[2][8];
#pragma unroll
  for (int t = 0; t < 2; t++)
#pragma unroll
    for (int r = 0; r < 8; r++) { mrun[t][r] = -1e30f; lrun[t][r] = 0.0f; }

  for (int j0 = 0; j0 < N_PAD; j0 += 32) {
    if (j0 + 32 < N_PAD) {
      __builtin_prefetch(Kb + (size_t)(j0 + 32) * DP, 0, 1);
      __builtin_prefetch(Vb + (size_t)(j0 + 32), 0, 1);
    }
    // scores for 32 key columns x 32 query rows: 4 WMMAs off 2 K fragments
    v16bf kb0 = load_B(Kb + (size_t)j0 * DP, DP, lane);
    v16bf kb1 = load_B(Kb + (size_t)(j0 + 16) * DP, DP, lane);
    v8f z = {};
    v8f s00 = wmma_bf16(qa0, kb0, z);
    v8f s01 = wmma_bf16(qa0, kb1, z);
    v8f s10 = wmma_bf16(qa1, kb0, z);
    v8f s11 = wmma_bf16(qa1, kb1, z);

    int m0c = j0 + nl, m1c = j0 + 16 + nl;
    bool v0 = (m0c < N_TOT), v1 = (m1c < N_TOT);
    float alpha[2][8];
#pragma unroll
    for (int t = 0; t < 2; t++) {
      v8f sa = t ? s10 : s00;
      v8f sb = t ? s11 : s01;
      float sm0[8], sm1[8], tmax[8];
#pragma unroll
      for (int r = 0; r < 8; r++) {
        sm0[r] = v0 ? sa[r] * tscale : -1e30f;
        sm1[r] = v1 ? sb[r] * tscale : -1e30f;
        tmax[r] = fmaxf(sm0[r], sm1[r]);
      }
      // row-wise max across the 16 lanes of each half-wave
#pragma unroll
      for (int off = 1; off < 16; off <<= 1)
#pragma unroll
        for (int r = 0; r < 8; r++)
          tmax[r] = fmaxf(tmax[r], __shfl_xor(tmax[r], off, 32));

      float p0[8], p1[8], psum[8];
#pragma unroll
      for (int r = 0; r < 8; r++) {
        float mnew = fmaxf(mrun[t][r], tmax[r]);
        alpha[t][r] = __expf(mrun[t][r] - mnew);
        mrun[t][r]  = mnew;
        p0[r] = __expf(sm0[r] - mnew);
        p1[r] = __expf(sm1[r] - mnew);
        psum[r] = p0[r] + p1[r];
      }
#pragma unroll
      for (int off = 1; off < 16; off <<= 1)
#pragma unroll
        for (int r = 0; r < 8; r++)
          psum[r] += __shfl_xor(psum[r], off, 32);
#pragma unroll
      for (int r = 0; r < 8; r++)
        lrun[t][r] = lrun[t][r] * alpha[t][r] + psum[r];

      // P -> LDS (C-layout lane=column  ==>  rows t*16 + r + 8*half)
#pragma unroll
      for (int r = 0; r < 8; r++) {
        Plds[(t * 16 + r + 8 * qh) * 32 + nl]      = (__bf16)p0[r];
        Plds[(t * 16 + r + 8 * qh) * 32 + 16 + nl] = (__bf16)p1[r];
      }
    }
    __syncthreads();
    v16bf pa0, pa1;
#pragma unroll
    for (int t = 0; t < 2; t++) {
      const __bf16* pr = &Plds[(t * 16 + (lane & 15)) * 32 + qh * 8];
      v8bf lo = *(const v8bf*)(pr);
      v8bf hi = *(const v8bf*)(pr + 16);
      v16bf pa;
#pragma unroll
      for (int i = 0; i < 8; i++) { pa[i] = lo[i]; pa[i + 8] = hi[i]; }
      if (t == 0) pa0 = pa; else pa1 = pa;
    }
    __syncthreads();

    // V fragments (Vt is [dp][n]: contiguous B-fragment loads), shared by both tiles
    v16bf vb0 = load_B(Vb + j0, N_PAD, lane);                       // d = 0..15
    v16bf vb1 = load_B(Vb + (size_t)16 * N_PAD + j0, N_PAD, lane);  // d = 16..31

#pragma unroll
    for (int r = 0; r < 8; r++) {
      o00[r] *= alpha[0][r]; o01[r] *= alpha[0][r];
      o10[r] *= alpha[1][r]; o11[r] *= alpha[1][r];
    }
    o00 = wmma_bf16(pa0, vb0, o00);
    o01 = wmma_bf16(pa0, vb1, o01);
    o10 = wmma_bf16(pa1, vb0, o10);
    o11 = wmma_bf16(pa1, vb1, o11);
  }

  // normalize and write attention output AO[b][n][h*24 + d] (bf16)
  int n0 = it * 32;
#pragma unroll
  for (int t = 0; t < 2; t++) {
    v8f oa = t ? o10 : o00;
    v8f ob = t ? o11 : o01;
#pragma unroll
    for (int r = 0; r < 8; r++) {
      float inv = 1.0f / lrun[t][r];
      int n = n0 + t * 16 + r + 8 * qh;
      size_t base = ((size_t)b * N_PAD + n) * C_ + (size_t)h * D_;
      AO[base + nl] = (__bf16)(oa[r] * inv);                    // d = nl (< 16)
      if (nl < 8) AO[base + 16 + nl] = (__bf16)(ob[r] * inv);   // d = 16 + nl (< 24)
    }
  }
}

// ---------------- output projection ----------------
// out[b][n][o] = sum_c AO[b][n][c] * Wout[o][c], n < 4096, fp32 output
// grid = (C_/16, B*N_X/16)
__global__ __launch_bounds__(32) void k_outproj(const __bf16* __restrict__ AO,
                                                const __bf16* __restrict__ wout,
                                                float* __restrict__ out) {
  int ot = blockIdx.x;
  int nt = blockIdx.y;              // over B*4096/16 = 512
  int lane = threadIdx.x;
  int b  = nt >> 8;                 // / 256
  int n0 = (nt & 255) * 16;

  const __bf16* a0 = AO   + ((size_t)b * N_PAD + n0) * C_;
  const __bf16* b0 = wout + (size_t)ot * 16 * C_;
  v8f acc = {};
#pragma unroll
  for (int k = 0; k < C_; k += 32) {
    acc = wmma_bf16(load_A(a0 + k, C_, lane), load_B(b0 + k, C_, lane), acc);
  }
  int nl = lane & 15, qh = lane >> 4;
  int o = ot * 16 + nl;
#pragma unroll
  for (int r = 0; r < 8; r++) {
    int n = n0 + r + 8 * qh;
    out[((size_t)b * N_X + n) * C_ + o] = acc[r];
  }
}

// ---------------- host launcher ----------------
extern "C" void kernel_launch(void* const* d_in, const int* in_sizes, int n_in,
                              void* d_out, int out_size, void* d_ws, size_t ws_size,
                              hipStream_t stream) {
  const float* X    = (const float*)d_in[0];
  const float* S    = (const float*)d_in[1];
  const float* Wqkv = (const float*)d_in[2];
  const float* Wout = (const float*)d_in[3];
  const float* temp = (const float*)d_in[4];
  float* out = (float*)d_out;

  __bf16* ws  = (__bf16*)d_ws;
  __bf16* xin = ws;
  __bf16* wq  = xin + XIN_ELEMS;
  __bf16* wo  = wq + WQKV_ELEMS;
  __bf16* Q   = wo + WOUT_ELEMS;
  __bf16* K   = Q + QKV_ELEMS;
  __bf16* Vt  = K + QKV_ELEMS;
  __bf16* AO  = Vt + QKV_ELEMS;

  // 1) pack inputs to bf16 (+ zero padding rows)
  {
    int g = (int)((XIN_ELEMS + 255) / 256);
    k_pack_x<<<g, 256, 0, stream>>>(X, S, xin);
  }
  {
    int g = (int)((WQKV_ELEMS + WOUT_ELEMS + 255) / 256);
    k_pack_w<<<g, 256, 0, stream>>>(Wqkv, Wout, wq, wo);
  }
  // 2) zero Q/K/Vt (head-dim padding 24..31 must be zero)
  {
    size_t n32 = (3 * QKV_ELEMS) / 2;   // bf16 pairs as u32
    int g = (int)((n32 + 255) / 256);
    k_zero<<<g, 256, 0, stream>>>((uint32_t*)Q, n32);
  }
  // 3) QKV projection
  {
    dim3 grid(C3 / 16, B_ * N_PAD / 16);   // 36 x 516
    k_qkv<<<grid, 32, 0, stream>>>(xin, wq, Q, K, Vt);
  }
  // 4) flash attention (two 16-row M-tiles per wave)
  {
    dim3 grid(N_PAD / 32, B_ * H_);        // 129 x 16
    k_attn<<<grid, 32, 0, stream>>>(Q, K, Vt, temp, AO);
  }
  // 5) output projection -> fp32 d_out
  {
    dim3 grid(C_ / 16, B_ * N_X / 16);     // 12 x 512
    k_outproj<<<grid, 32, 0, stream>>>(AO, wo, out);
  }
}
